// QCNNModel_89893665505557
// MI455X (gfx1250) — compile-verified, matrix operand order checked by reading, not run on hardware
//
#include <hip/hip_runtime.h>
#include <hip/hip_bf16.h>
#include <math.h>

// CDNA5 WMMA vector types (probe-verified signatures)
typedef __attribute__((ext_vector_type(16))) _Float16 v16h;
typedef __attribute__((ext_vector_type(8)))  float    v8f;
typedef __attribute__((ext_vector_type(4)))  int      v4i;

#define NQ   8
#define DIM  256      // 2^NQ amplitudes
#define TPB  256      // threads per block (8 wave32)
#define SPB  64       // samples per block in main kernel

// gfx1250 async LDS-DMA path (ASYNCcnt). Guarded: falls back to direct
// global loads if the toolchain does not expose the builtins.
#if __has_builtin(__builtin_amdgcn_global_load_async_to_lds_b128) && \
    __has_builtin(__builtin_amdgcn_s_wait_asynccnt)
#define HAVE_ASYNC_LDS 1
typedef __attribute__((address_space(1))) v4i* gv4ip;   // global (prints __device__)
typedef __attribute__((address_space(3))) v4i* lv4ip;   // LDS    (prints __shared__)
#else
#define HAVE_ASYNC_LDS 0
#endif

// ---------------------------------------------------------------------------
// ISA 7.12.2: 16-bit A-operand 16x32 per-lane K mapping.
// lane 0-15 : VGPR0-3 -> K0..7,  VGPR4-7 -> K16..23   (2 elems per VGPR)
// lane 16-31: VGPR0-3 -> K8..15, VGPR4-7 -> K24..31
// element e (0..15) of the v16h:  K = (e&7) + ((e>>3)<<4) + ((lane>>4)<<3)
// B-operand mirrored (N = lane&15, same K packing).
// ---------------------------------------------------------------------------
__device__ __forceinline__ int kmap(int lane, int e) {
    return (e & 7) + ((e >> 3) << 4) + ((lane >> 4) << 3);
}

// ======================= gate helpers (prep kernel 1) ======================
// state tile: sr/si [256 amps][16 cols] in LDS. wire w <-> bit (7-w) of amp.

__device__ __forceinline__ void gate_ry(float (*sr)[16], float (*si)[16],
                                        int w, float th) {
    float s, c; __sincosf(th * 0.5f, &s, &c);
    const int bw = (NQ - 1) - w;
    for (int f = threadIdx.x; f < 128 * 16; f += TPB) {
        int jl = f & 15, p = f >> 4;
        int i0 = ((p >> bw) << (bw + 1)) | (p & ((1 << bw) - 1));
        int i1 = i0 | (1 << bw);
        float a0r = sr[i0][jl], a0i = si[i0][jl];
        float a1r = sr[i1][jl], a1i = si[i1][jl];
        sr[i0][jl] = c * a0r - s * a1r;  si[i0][jl] = c * a0i - s * a1i;
        sr[i1][jl] = s * a0r + c * a1r;  si[i1][jl] = s * a0i + c * a1i;
    }
}

__device__ __forceinline__ void gate_rz(float (*sr)[16], float (*si)[16],
                                        int w, float th) {
    float s, c; __sincosf(th * 0.5f, &s, &c);
    const int bw = (NQ - 1) - w;
    for (int f = threadIdx.x; f < 128 * 16; f += TPB) {
        int jl = f & 15, p = f >> 4;
        int i0 = ((p >> bw) << (bw + 1)) | (p & ((1 << bw) - 1));
        int i1 = i0 | (1 << bw);
        float a0r = sr[i0][jl], a0i = si[i0][jl];
        float a1r = sr[i1][jl], a1i = si[i1][jl];
        // |0>: *(c - i s)   |1>: *(c + i s)
        sr[i0][jl] = a0r * c + a0i * s;  si[i0][jl] = a0i * c - a0r * s;
        sr[i1][jl] = a1r * c - a1i * s;  si[i1][jl] = a1i * c + a1r * s;
    }
}

__device__ __forceinline__ void gate_cnot(float (*sr)[16], float (*si)[16],
                                          int cq) {
    const int bc = (NQ - 1) - cq;   // control bit
    const int bt = bc - 1;          // target bit (wire cq+1)
    for (int f = threadIdx.x; f < 64 * 16; f += TPB) {
        int jl = f & 15, p = f >> 4;                 // 6 free bits
        int lo = p & ((1 << bt) - 1);
        int hi = p >> bt;
        int i0 = (hi << (bc + 1)) | (1 << bc) | lo;  // control=1, target=0
        int i1 = i0 | (1 << bt);                     // control=1, target=1
        float r = sr[i0][jl], im = si[i0][jl];
        sr[i0][jl] = sr[i1][jl]; si[i0][jl] = si[i1][jl];
        sr[i1][jl] = r;          si[i1][jl] = im;
    }
}

// ============ Prep 1: fixed circuit unitary W = U (256x256 cplx) ===========
__global__ __launch_bounds__(TPB)
void build_unitary(const float* __restrict__ wc, const float* __restrict__ wp,
                   float* __restrict__ Wr, float* __restrict__ Wi) {
    __shared__ float sr[DIM][16];
    __shared__ float si[DIM][16];
    const int t = threadIdx.x;
    const int colbase = blockIdx.x * 16;

    for (int f = t; f < DIM * 16; f += TPB) {
        int amp = f >> 4, jl = f & 15;
        sr[amp][jl] = (amp == colbase + jl) ? 1.0f : 0.0f;
        si[amp][jl] = 0.0f;
    }
    __syncthreads();

    for (int i = 0; i < NQ; ++i) {
        gate_ry(sr, si, i, wc[0 * NQ + i]); __syncthreads();
        gate_rz(sr, si, i, wc[1 * NQ + i]); __syncthreads();
    }
    for (int i = 0; i < NQ - 1; ++i) {
        gate_cnot(sr, si, i); __syncthreads();
    }
    for (int i = 0; i < NQ; ++i) {
        gate_ry(sr, si, i, wc[2 * NQ + i]); __syncthreads();
        gate_rz(sr, si, i, wc[3 * NQ + i]); __syncthreads();
    }
    for (int i = 0; i < NQ / 2; ++i) {
        gate_ry(sr, si, i, wp[i]); __syncthreads();
    }

    for (int f = t; f < DIM * 16; f += TPB) {
        int amp = f >> 4, jl = f & 15;
        Wr[amp * DIM + colbase + jl] = sr[amp][jl];
        Wi[amp * DIM + colbase + jl] = si[amp][jl];
    }
}

// ==== Prep 2: A = Re(U^H (Z0) U), stored pre-swizzled as WMMA B-frags ======
__global__ __launch_bounds__(TPB)
void build_obs(const float* __restrict__ Wr, const float* __restrict__ Wi,
               _Float16* __restrict__ Afrag) {
    const int o    = blockIdx.x * TPB + threadIdx.x;   // 65536 elements
    const int e    = o & 15;
    const int lane = (o >> 4) & 31;
    const int nb   = (o >> 9) & 15;
    const int kb   = (o >> 13) & 7;
    const int j    = kb * 32 + kmap(lane, e);
    const int n    = nb * 16 + (lane & 15);
    float acc = 0.0f;
    for (int i = 0; i < 128; ++i)          // qubit-0 bit (MSB) == 0 -> +1
        acc += Wr[i * DIM + j] * Wr[i * DIM + n] + Wi[i * DIM + j] * Wi[i * DIM + n];
    for (int i = 128; i < 256; ++i)        // qubit-0 bit == 1 -> -1
        acc -= Wr[i * DIM + j] * Wr[i * DIM + n] + Wi[i * DIM + j] * Wi[i * DIM + n];
    Afrag[o] = (_Float16)acc;
}

#if HAVE_ASYNC_LDS
// Issue this thread's share (4 x B128) of one 16 KB K-slice copy:
// Afrag[kb] (global, L2-resident) -> bst (LDS), tracked by ASYNCcnt.
__device__ __forceinline__ void stage_slice(const _Float16* __restrict__ Afrag,
                                            _Float16* bst, int kb) {
    const char* g  = (const char*)Afrag + kb * 16384;
    unsigned    lb = (unsigned)(uintptr_t)bst;   // generic LDS ptr low 32 bits
#pragma unroll
    for (int i = 0; i < 4; ++i) {
        int c = i * TPB + (int)threadIdx.x;      // 1024 chunks of 16 B
        __builtin_amdgcn_global_load_async_to_lds_b128(
            (gv4ip)(uintptr_t)(g + c * 16),
            (lv4ip)(unsigned)(lb + c * 16),
            0, 0);
    }
}
#endif

// ================= Main: V-frag gen + WMMA GEMM + dot + MLP ================
__global__ __launch_bounds__(TPB)
void qcnn_main(const float* __restrict__ x, const _Float16* __restrict__ Afrag,
               const float* __restrict__ W1, const float* __restrict__ b1,
               const float* __restrict__ W2, const float* __restrict__ b2,
               float* __restrict__ out, int B) {
    __shared__ float cs[SPB][2 * NQ];                        // 4 KB
    __shared__ __align__(32) _Float16 vfrag[4][8][32][16];   // 32 KB A-operands
#if HAVE_ASYNC_LDS
    __shared__ __align__(32) _Float16 bstage[2][16][32][16]; // 2 x 16 KB B-frags
#endif
    __shared__ float zpart[SPB];

    const int t    = threadIdx.x;
    const int base = blockIdx.x * SPB;

    if (t < SPB) zpart[t] = 0.0f;

    // ---- per-sample cos/sin(x/2) table --------------------------------
    for (int f = t; f < SPB * NQ; f += TPB) {
        int s = f >> 3, w = f & 7;
        int g = base + s; if (g >= B) g = B - 1;
        float sn, c; __sincosf(x[g * NQ + w] * 0.5f, &sn, &c);
        cs[s][2 * w]     = c;
        cs[s][2 * w + 1] = sn;
    }
    __syncthreads();

    // ---- build V fragments directly in A-operand swizzle --------------
    // vfrag[r][kb][lane][e] = v[sample r*16+(lane&15)][j = kb*32 + kmap(lane,e)]
    for (int f = t; f < 4 * 8 * 32 * 16; f += TPB) {
        int e = f & 15, l = (f >> 4) & 31, kb = (f >> 9) & 7, r = (f >> 12) & 3;
        int s = r * 16 + (l & 15);
        int j = kb * 32 + kmap(l, e);
        float prod = 1.0f;
#pragma unroll
        for (int w = 0; w < NQ; ++w) {
            int bit = (j >> ((NQ - 1) - w)) & 1;    // wire w <-> bit 7-w
            prod *= cs[s][2 * w + bit];
        }
        vfrag[r][kb][l][e] = (_Float16)prod;
    }
    __syncthreads();

    // ---- GEMM C = V * A via v_wmma_f32_16x16x32_f16 -------------------
    const int lane = t & 31;
    const int wv   = t >> 5;        // wave id 0..7
    const int r    = wv & 3;        // 16-sample row tile
    const int nb0  = (wv >> 2) * 8; // 8 column tiles per wave

    v8f zero8 = {};
    v8f acc[8];
#pragma unroll
    for (int q = 0; q < 8; ++q) acc[q] = zero8;

#if HAVE_ASYNC_LDS
    // double-buffered async staging of A K-slices: copy kb+1 overlaps wmma(kb)
    stage_slice(Afrag, &bstage[0][0][0][0], 0);
#endif
    for (int kb = 0; kb < 8; ++kb) {
#if HAVE_ASYNC_LDS
        const int buf = kb & 1;
        if (kb < 7) {
            stage_slice(Afrag, &bstage[buf ^ 1][0][0][0], kb + 1);
            __builtin_amdgcn_s_wait_asynccnt(4);  // slice kb done (in-order)
        } else {
            __builtin_amdgcn_s_wait_asynccnt(0);
        }
        __syncthreads();   // all waves' copies of slice kb visible
#endif
        v16h a = *(const v16h*)(&vfrag[r][kb][lane][0]);
#if !HAVE_ASYNC_LDS
        if (kb < 7)
            __builtin_prefetch(Afrag + ((((kb + 1) * 16) + nb0) * 32 + lane) * 16, 0, 0);
#endif
#pragma unroll
        for (int q = 0; q < 8; ++q) {
#if HAVE_ASYNC_LDS
            v16h bm = *(const v16h*)(&bstage[buf][nb0 + q][lane][0]);
#else
            v16h bm = *(const v16h*)(Afrag + (((kb * 16) + nb0 + q) * 32 + lane) * 16);
#endif
            acc[q] = __builtin_amdgcn_wmma_f32_16x16x32_f16(
                false, a, false, bm, (short)0, acc[q], false, false);
        }
#if HAVE_ASYNC_LDS
        __syncthreads();   // buffer `buf` is re-staged at iteration kb+1
#endif
    }

    // ---- z0[s] = sum_n C[s][n] * v[s][n]  (C/D layout: M=p+8*(lane>=16),
    //      N=lane&15); xor-tree over bits 0..3 keeps M-halves separate ----
    const int mhalf = (lane >> 4) << 3;   // 0 or 8
#pragma unroll
    for (int p = 0; p < 8; ++p) {
        int m = p + mhalf;
        float tp = 0.0f;
#pragma unroll
        for (int q = 0; q < 8; ++q) {
            int n   = (nb0 + q) * 16 + (lane & 15);
            int kb2 = n >> 5;
            int K2  = n & 31;
            int l2  = (((K2 >> 3) & 1) << 4) | m;       // invert kmap
            int e2  = (K2 & 7) | ((K2 >> 4) << 3);
            tp += acc[q][p] * (float)vfrag[r][kb2][l2][e2];
        }
        tp += __shfl_xor(tp, 1, 32);
        tp += __shfl_xor(tp, 2, 32);
        tp += __shfl_xor(tp, 4, 32);
        tp += __shfl_xor(tp, 8, 32);
        if ((lane & 15) == 0) atomicAdd(&zpart[r * 16 + m], tp);
    }
    __syncthreads();

    // ---- fused MLP head: sigmoid(W2 . relu(z*W1 + b1) + b2) -----------
    if (t < SPB) {
        int g = base + t;
        if (g < B) {
            float z = zpart[t];
            float o = b2[0];
#pragma unroll
            for (int jj = 0; jj < 16; ++jj) {
                float h = z * W1[jj] + b1[jj];
                h = h > 0.0f ? h : 0.0f;
                o += W2[jj] * h;
            }
            out[g] = 1.0f / (1.0f + __expf(-o));
        }
    }
}

// ===========================================================================
extern "C" void kernel_launch(void* const* d_in, const int* in_sizes, int n_in,
                              void* d_out, int out_size, void* d_ws, size_t ws_size,
                              hipStream_t stream) {
    const float* x  = (const float*)d_in[0];
    const float* wc = (const float*)d_in[1];
    const float* wp = (const float*)d_in[2];
    const float* W1 = (const float*)d_in[3];
    const float* b1 = (const float*)d_in[4];
    const float* W2 = (const float*)d_in[5];
    const float* b2 = (const float*)d_in[6];
    float* out = (float*)d_out;
    const int B = in_sizes[0] / NQ;

    // workspace: Wr (256 KB) | Wi (256 KB) | Afrag f16 swizzled (128 KB)
    float*    Wr    = (float*)d_ws;
    float*    Wi    = Wr + DIM * DIM;
    _Float16* Afrag = (_Float16*)(Wi + DIM * DIM);

    build_unitary<<<16, TPB, 0, stream>>>(wc, wp, Wr, Wi);
    build_obs<<<DIM * DIM / TPB, TPB, 0, stream>>>(Wr, Wi, Afrag);
    qcnn_main<<<(B + SPB - 1) / SPB, TPB, 0, stream>>>(x, Afrag, W1, b1, W2, b2,
                                                       out, B);
}
